// MultiHeadAttention_18262200943441
// MI455X (gfx1250) — matrix operand order repaired and checked by reading, not compiled, not run on hardware
//
#include <hip/hip_runtime.h>

// ---------------- CDNA5 WMMA types / helpers (gfx1250, wave32) ----------------
typedef __attribute__((ext_vector_type(16))) __bf16 v16bf;
typedef __attribute__((ext_vector_type(8)))  float  v8f;
typedef __attribute__((ext_vector_type(4)))  unsigned int v4u;
typedef __attribute__((ext_vector_type(8)))  unsigned int v8u;

#define WMMA_BF16(a, b, c) \
  __builtin_amdgcn_wmma_f32_16x16x32_bf16(false, (a), false, (b), (short)0, (c), false, false)

__device__ inline v8f vzero8() {
  v8f z = {0.f, 0.f, 0.f, 0.f, 0.f, 0.f, 0.f, 0.f};
  return z;
}

constexpr int  Bc = 32, Nc = 325, Lc = 12, Dc = 128, Hc = 8;
constexpr int  BN = Bc * Nc;                       // 10400 (b,n) tiles
constexpr long OUT1 = (long)BN * Lc * Dc;          // 15,974,400 floats
constexpr long ARET_STRIDE = (long)BN * Hc * Lc;   // 998,400 (stride of q in attn_ret)
constexpr int  TILE = 16 * Dc;                     // padded 16x128 tile (2048 elems)
constexpr int  ROWBYTES = Lc * Dc;                 // 1536 valid elements per tile
constexpr float BN_CNT = 124800.0f;                // B*N*L

// ---------------- Tensor Data Mover: 1-row 1536-elem fp32 tile, global -> LDS ----------
// D# per CDNA5 ISA. Groups 2/3 omitted (<=2D tensor).
__device__ inline void tdm_load_tile(const float* gsrc, unsigned lds_off) {
  const unsigned long long ga = (unsigned long long)(uintptr_t)gsrc;
  v4u g0;
  g0[0] = 1u;                                             // count=1, user mode, no gather
  g0[1] = lds_off;                                        // lds_addr (bytes)
  g0[2] = (unsigned)(ga & 0xFFFFFFFFu);                   // global_addr[31:0]
  g0[3] = (unsigned)((ga >> 32) & 0x01FFFFFFu)            // global_addr[56:32]
          | (2u << 30);                                   // type = 2 ("image")
  v8u g1;
  g1[0] = 2u << 16;                                       // data_size = 4 bytes
  g1[1] = (unsigned)(ROWBYTES & 0xFFFF) << 16;            // tensor_dim0[15:0] (bits 63:48)
  g1[2] = (unsigned)(ROWBYTES >> 16) | (1u << 16);        // tensor_dim0[31:16], tensor_dim1=1
  g1[3] = (unsigned)ROWBYTES << 16;                       // tile_dim0 = 1536
  g1[4] = 1u;                                             // tile_dim1 = 1
  g1[5] = (unsigned)ROWBYTES;                             // tensor_dim0_stride = 1536
  g1[6] = 0u;
  g1[7] = 0u;
  asm volatile("tensor_load_to_lds %0, %1" :: "s"(g0), "s"(g1) : "memory");
}

// A fragment (16x32, bf16) from row-major bf16 LDS tile [16][128].
// A k order used here (contiguous, same as B): lane l -> row l%16; K = (l&16?16:0) + e
__device__ inline v16bf frag_A_bf16(const __bf16* s, int lane, int kbase) {
  const int row = lane & 15;
  const int kh  = (lane & 16) ? 16 : 0;
  const __bf16* p = s + row * Dc + kbase + kh;
  v16bf a;
#pragma unroll
  for (int e = 0; e < 8; ++e) a[e] = p[e];
#pragma unroll
  for (int e = 0; e < 8; ++e) a[8 + e] = p[8 + e];
  return a;
}

// A fragment (16x32, bf16) from row-major fp32 LDS tile [16][128] (convert on build).
__device__ inline v16bf frag_A_f32(const float* s, int lane, int kbase) {
  const int row = lane & 15;
  const int kh  = (lane & 16) ? 16 : 0;
  const float* p = s + row * Dc + kbase + kh;
  v16bf a;
#pragma unroll
  for (int e = 0; e < 8; ++e) a[e] = (__bf16)p[e];
#pragma unroll
  for (int e = 0; e < 8; ++e) a[8 + e] = (__bf16)p[8 + e];
  return a;
}

// B fragment (32x16, bf16) built as B[k][n] = W[n][k] from row-major bf16 weight W[128][128].
// B k order used here (contiguous, same as A): lane l -> col N = l%16; K = (l&16?16:0) + e
__device__ inline v16bf frag_B_w(const __bf16* w, int lane, int nbase, int kbase) {
  const __bf16* p = w + (nbase + (lane & 15)) * Dc + kbase + ((lane & 16) ? 16 : 0);
  v16bf b;
#pragma unroll
  for (int e = 0; e < 16; ++e) b[e] = p[e];
  return b;
}

// A fragment with only K<16 valid (dk=16 padded to K=32 with zeros), fp32 source.
__device__ inline v16bf frag_A_f32k16(const float* s, int pitch, int lane, int colbase) {
  const int row = lane & 15;
  v16bf a;
#pragma unroll
  for (int e = 0; e < 16; ++e) a[e] = (lane & 16) ? (__bf16)0.0f : (__bf16)s[row * pitch + colbase + e];
  return a;
}

// B fragment for Q*K^T: B[k=d][n=keypos] = K[keypos][colbase+d]; d>=16 lanes are zero.
__device__ inline v16bf frag_BT_f32(const float* s, int lane, int colbase) {
  v16bf b;
  if (lane & 16) {
#pragma unroll
    for (int e = 0; e < 16; ++e) b[e] = (__bf16)0.0f;
  } else {
    const int kp = lane & 15;  // key position (rows >= 12 are exact zeros)
#pragma unroll
    for (int e = 0; e < 16; ++e) b[e] = (__bf16)s[kp * Dc + colbase + e];
  }
  return b;
}

// B fragment for attn*V: B[k=keypos][n=d] = V[keypos][colbase+d]; keypos>=16 lanes are zero.
__device__ inline v16bf frag_Bcol_f32(const float* s, int lane, int colbase) {
  v16bf b;
  if (lane & 16) {
#pragma unroll
    for (int e = 0; e < 16; ++e) b[e] = (__bf16)0.0f;
  } else {
    const int d = lane & 15;
#pragma unroll
    for (int e = 0; e < 16; ++e) b[e] = (__bf16)s[e * Dc + colbase + d];
  }
  return b;
}

// ---------------- kernel 0: weight-norm -> bf16 weights ----------------
__global__ __launch_bounds__(128) void k_wnorm(const float* __restrict__ v,
                                               const float* __restrict__ g,
                                               __bf16* __restrict__ out) {
  __shared__ float red[128];
  const int row = blockIdx.x, t = threadIdx.x;
  const float x = v[row * Dc + t];
  red[t] = x * x;
  __syncthreads();
#pragma unroll
  for (int s = 64; s > 0; s >>= 1) {
    if (t < s) red[t] += red[t + s];
    __syncthreads();
  }
  const float inv = g[row] * rsqrtf(red[0]);   // g / ||v||_row
  out[row * Dc + t] = (__bf16)(x * inv);
}

// ---------------- kernel 1: fused QKV proj + rel-pos attention + context ----------------
__global__ __launch_bounds__(256) void k_attn(
    const float* __restrict__ inQ, const float* __restrict__ inK, const float* __restrict__ inV,
    const __bf16* __restrict__ wqb, const __bf16* __restrict__ wkb, const __bf16* __restrict__ wvb,
    const float* __restrict__ relk, const float* __restrict__ relv,
    const float* __restrict__ w1g, const float* __restrict__ w2g,
    float* __restrict__ ctx, float* __restrict__ out2) {
  // sStage: TDM staging for inQ/inK/inV (fp32); reused afterwards as sS/sA/sA2 (8 KB each)
  __shared__ float sStage[3][TILE];                              // 24 KB
  __shared__ float sQKV[3][TILE];                                // 24 KB (Q,K,V fp32)
  __shared__ float srelk[23 * 16], srelv[23 * 16], sw1[64], sw2[64];

  const int  t = threadIdx.x, lane = t & 31, wave = t >> 5;
  const long base = (long)blockIdx.x * ROWBYTES;

  // wave 0 kicks off 3 TDM transfers (EXEC-independent, tracked by TENSORcnt)
  if (wave == 0) {
    tdm_load_tile(inQ + base, (unsigned)(uintptr_t)(&sStage[0][0]));
    tdm_load_tile(inK + base, (unsigned)(uintptr_t)(&sStage[1][0]));
    tdm_load_tile(inV + base, (unsigned)(uintptr_t)(&sStage[2][0]));
  }
  // meanwhile: zero the pad rows 12..15 (disjoint from TDM destination) + small tables
#pragma unroll
  for (int p = 0; p < 3; ++p)
    for (int i = ROWBYTES + t; i < TILE; i += 256) sStage[p][i] = 0.f;
  for (int i = t; i < 23 * 16; i += 256) { srelk[i] = relk[i]; srelv[i] = relv[i]; }
  if (t < 64) { sw1[t] = w1g[t]; sw2[t] = w2g[t]; }
  if (wave == 0) __builtin_amdgcn_s_wait_tensorcnt(0);
  __syncthreads();

  // Q/K/V projections: wave w owns output columns [16w, 16w+16); 3 x 4 chained WMMAs
  {
    const __bf16* Ws[3] = { wqb, wkb, wvb };
    const int nb = wave * 16;
#pragma unroll
    for (int p = 0; p < 3; ++p) {
      v8f acc = vzero8();
#pragma unroll
      for (int kt = 0; kt < 4; ++kt) {
        v16bf a = frag_A_f32(&sStage[p][0], lane, kt * 32);   // fp32 -> bf16 on build
        v16bf b = frag_B_w(Ws[p], lane, nb, kt * 32);
        acc = WMMA_BF16(a, b, acc);
      }
      const int col = nb + (lane & 15);
      const int rb  = (lane & 16) ? 8 : 0;
      float* o = &sQKV[p][0];
#pragma unroll
      for (int j = 0; j < 8; ++j) o[(rb + j) * Dc + col] = acc[j];
    }
  }
  __syncthreads();

  // staging consumed -> alias the buffers for the attention stages
  float* sQ  = &sQKV[0][0];
  float* sK  = &sQKV[1][0];
  float* sV  = &sQKV[2][0];
  float* sS  = &sStage[0][0];   // scores            [8][16][16]
  float* sA  = &sStage[1][0];   // softmax(attn)     [8][16][16]
  float* sA2 = &sStage[2][0];   // w2-mixed attn     [8][16][16]

  // scores: s1 = Q_h K_h^T via one WMMA per head; s2 (rel-pos keys) on VALU; scale=1/sqrt(16)
  {
    const int h = wave;
    v16bf a = frag_A_f32k16(sQ, Dc, lane, h * 16);
    v16bf b = frag_BT_f32(sK, lane, h * 16);
    v8f  s  = WMMA_BF16(a, b, vzero8());
    const int kc = lane & 15;
    const int rb = (lane & 16) ? 8 : 0;
#pragma unroll
    for (int j = 0; j < 8; ++j) {
      const int q = rb + j;
      float val = 0.f;
      if (q < Lc && kc < Lc) {
        float s2 = 0.f;
        const int r = kc - q + 11;          // clip(k-q,-11,11)+11; |k-q|<=11 for L=12
#pragma unroll
        for (int d = 0; d < 16; ++d) s2 += sQ[q * Dc + h * 16 + d] * srelk[r * 16 + d];
        val = (s[j] + s2) * 0.25f;
      }
      sS[h * 256 + q * 16 + kc] = val;
    }
  }
  __syncthreads();

  // cross-head mix (w1) + leaky-ReLU + softmax over k
  if (t < 128) {
    const int g = t & 7, q = t >> 3;
    if (q < Lc) {
      float row[Lc];
      float mx = -3.0e38f;
#pragma unroll
      for (int k = 0; k < Lc; ++k) {
        float m = 0.f;
#pragma unroll
        for (int h = 0; h < 8; ++h) m += sS[h * 256 + q * 16 + k] * sw1[h * 8 + g];
        m = (m > 0.f) ? m : 0.2f * m;
        row[k] = m;
        mx = fmaxf(mx, m);
      }
      float sum = 0.f;
#pragma unroll
      for (int k = 0; k < Lc; ++k) { const float e = __expf(row[k] - mx); row[k] = e; sum += e; }
      const float inv = 1.0f / sum;
#pragma unroll
      for (int k = 0; k < Lc; ++k) sA[g * 256 + q * 16 + k] = row[k] * inv;
#pragma unroll
      for (int k = Lc; k < 16; ++k) sA[g * 256 + q * 16 + k] = 0.f;
    } else {
#pragma unroll
      for (int k = 0; k < 16; ++k) sA[g * 256 + q * 16 + k] = 0.f;
    }
  }
  __syncthreads();

  // cross-head mix (w2); emit attn_ret[q][(bn*H+g)][k]
  if (t < 128) {
    const int g = t & 7, q = t >> 3;
    float* aout = out2 + (long)q * ARET_STRIDE + ((long)blockIdx.x * Hc + g) * Lc;
#pragma unroll
    for (int k = 0; k < 16; ++k) {
      float m = 0.f;
      if (q < Lc && k < Lc) {
#pragma unroll
        for (int h = 0; h < 8; ++h) m += sA[h * 256 + q * 16 + k] * sw2[h * 8 + g];
        aout[k] = m;
      }
      sA2[g * 256 + q * 16 + k] = m;
    }
  }
  __syncthreads();

  // context: c1 = attn @ V_h (one WMMA per head) + c2 = attn @ relV (VALU)
  {
    const int g = wave;
    v16bf a = frag_A_f32k16(sA2 + g * 256, 16, lane, 0);
    v16bf b = frag_Bcol_f32(sV, lane, g * 16);
    v8f  c  = WMMA_BF16(a, b, vzero8());
    const int d  = lane & 15;
    const int rb = (lane & 16) ? 8 : 0;
    float* sCtx = sK;  // reuse: K fully consumed
#pragma unroll
    for (int j = 0; j < 8; ++j) {
      const int q = rb + j;
      float c2 = 0.f;
      if (q < Lc) {
#pragma unroll
        for (int k = 0; k < Lc; ++k)
          c2 += sA2[g * 256 + q * 16 + k] * srelv[(k - q + 11) * 16 + d];
      }
      sCtx[q * Dc + g * 16 + d] = c[j] + c2;
    }
  }
  __syncthreads();

  // coalesced ctx spill (fp32) for the grid-wide BatchNorm pass
  {
    float* cg = ctx + base;
    const float* sCtx = sK;
    for (int i = t; i < ROWBYTES; i += 256) cg[i] = sCtx[i];
  }
}

// ---------------- kernel 2a: deterministic BN partial sums (416 blocks x 25 tiles) --------
__global__ __launch_bounds__(256) void k_bnpart(const float* __restrict__ ctx,
                                                float* __restrict__ part) {
  const int p = blockIdx.x, t = threadIdx.x;
  const float* c = ctx + (long)p * (25 * ROWBYTES);
  float s = 0.f, ss = 0.f;
  for (int i = t; i < 25 * ROWBYTES; i += 256) {  // stride 256 keeps channel = t & 127 fixed
    const float v = c[i];
    s += v; ss += v * v;
  }
  __shared__ float rs[256], rss[256];
  rs[t] = s; rss[t] = ss;
  __syncthreads();
  if (t < 128) {
    part[p * 256 + t]       = rs[t] + rs[t + 128];
    part[p * 256 + 128 + t] = rss[t] + rss[t + 128];
  }
}

// ---------------- kernel 2b: finalize mean / gamma*rsqrt(var+eps) ----------------
__global__ __launch_bounds__(128) void k_bnfinal(const float* __restrict__ part,
                                                 const float* __restrict__ gamma,
                                                 float* __restrict__ mean,
                                                 float* __restrict__ scale) {
  const int t = threadIdx.x;
  float s = 0.f, ss = 0.f;
  for (int p = 0; p < 416; ++p) { s += part[p * 256 + t]; ss += part[p * 256 + 128 + t]; }
  const float inv = 1.0f / BN_CNT;
  const float m = s * inv;
  const float var = ss * inv - m * m;
  mean[t]  = m;
  scale[t] = gamma[t] * rsqrtf(var + 1e-5f);
}

// ---------------- kernel 3: normalize + fc GEMM + ReLU + residual ----------------
__global__ __launch_bounds__(256) void k_out(
    const float* __restrict__ ctx, const float* __restrict__ mean,
    const float* __restrict__ scale, const float* __restrict__ beta,
    const __bf16* __restrict__ wfc, const float* __restrict__ inV,
    float* __restrict__ outp) {
  __shared__ __bf16 sX[TILE];
  const int  t = threadIdx.x, lane = t & 31, wave = t >> 5;
  const long base = (long)blockIdx.x * ROWBYTES;
  const float* c = ctx + base;
  for (int i = t; i < TILE; i += 256) {
    const int col = i & 127;
    float v = 0.f;
    if (i < ROWBYTES) v = (c[i] - mean[col]) * scale[col] + beta[col];
    sX[i] = (__bf16)v;
  }
  __syncthreads();
  v8f acc = vzero8();
  const int nb = wave * 16;
#pragma unroll
  for (int kt = 0; kt < 4; ++kt) {
    v16bf a = frag_A_bf16(sX, lane, kt * 32);
    v16bf b = frag_B_w(wfc, lane, nb, kt * 32);
    acc = WMMA_BF16(a, b, acc);
  }
  const int col = nb + (lane & 15);
  const int rb  = (lane & 16) ? 8 : 0;
#pragma unroll
  for (int j = 0; j < 8; ++j) {
    const int q = rb + j;
    if (q < Lc) {
      const long idx = base + q * Dc + col;
      outp[idx] = fmaxf(acc[j], 0.f) + inV[idx];
    }
  }
}

// ---------------- launch ----------------
extern "C" void kernel_launch(void* const* d_in, const int* in_sizes, int n_in,
                              void* d_out, int out_size, void* d_ws, size_t ws_size,
                              hipStream_t stream) {
  (void)in_sizes; (void)n_in; (void)out_size; (void)ws_size;
  const float* inQ   = (const float*)d_in[0];
  const float* inK   = (const float*)d_in[1];
  const float* inV   = (const float*)d_in[2];
  const float* wq_v  = (const float*)d_in[3];
  const float* wq_g  = (const float*)d_in[4];
  const float* wk_v  = (const float*)d_in[5];
  const float* wk_g  = (const float*)d_in[6];
  const float* wv_v  = (const float*)d_in[7];
  const float* wv_g  = (const float*)d_in[8];
  const float* fc_v  = (const float*)d_in[9];
  const float* fc_g  = (const float*)d_in[10];
  const float* relk  = (const float*)d_in[11];
  const float* relv  = (const float*)d_in[12];
  const float* w1    = (const float*)d_in[13];
  const float* w2    = (const float*)d_in[14];
  const float* gamma = (const float*)d_in[15];
  const float* beta  = (const float*)d_in[16];

  char* ws = (char*)d_ws;
  __bf16* wqb     = (__bf16*)(ws + 0);        // 4 x 32 KB bf16 weights (L2-resident)
  __bf16* wkb     = (__bf16*)(ws + 32768);
  __bf16* wvb     = (__bf16*)(ws + 65536);
  __bf16* wfcb    = (__bf16*)(ws + 98304);
  float*  bnmean  = (float*)(ws + 131072);
  float*  bnscale = (float*)(ws + 131584);
  float*  ctx     = (float*)(ws + 132096);    // B*N*L*D fp32 = 63.9 MB
  float*  part    = (float*)(ws + 132096 + (size_t)BN * Lc * Dc * 4);  // 416*256 floats

  float* out1 = (float*)d_out;
  float* out2 = out1 + OUT1;

  k_wnorm<<<128, 128, 0, stream>>>(wq_v, wq_g, wqb);
  k_wnorm<<<128, 128, 0, stream>>>(wk_v, wk_g, wkb);
  k_wnorm<<<128, 128, 0, stream>>>(wv_v, wv_g, wvb);
  k_wnorm<<<128, 128, 0, stream>>>(fc_v, fc_g, wfcb);
  k_attn<<<BN, 256, 0, stream>>>(inQ, inK, inV, wqb, wkb, wvb, relk, relv, w1, w2, ctx, out2);
  k_bnpart<<<416, 256, 0, stream>>>(ctx, part);
  k_bnfinal<<<1, 128, 0, stream>>>(part, gamma, bnmean, bnscale);
  k_out<<<BN, 256, 0, stream>>>(ctx, bnmean, bnscale, beta, wfcb, inV, out1);
}
